// MultiHeadAttention_43387759624452
// MI455X (gfx1250) — compile-verified, matrix operand order checked by reading, not run on hardware
//
#include <hip/hip_runtime.h>
#include <hip/hip_bf16.h>

typedef __bf16 bf16_t;
typedef __attribute__((ext_vector_type(16))) __bf16 v16bf;
typedef __attribute__((ext_vector_type(8)))  float  v8f;

#define NEG_INF_MASK -1000000000.0f

struct FragBF {
    union {
        v16bf  v;
        bf16_t e[16];
        uint4  q[2];
    };
};

union PackBF8 {
    uint4  q;
    bf16_t e[8];
};

// 16-bit A (16x32) / B (32x16) fragment, both loaded from row-major memory:
// lane: r = lane&15 (A-row / B-column), half = lane>>4, kbase = 8*half.
// elements 0..7  -> K = kbase + j ; elements 8..15 -> K = kbase + 16 + (j-8).
// caller passes p = base + r*ld + kchunk + kbase ; chunks are 16B aligned.
__device__ __forceinline__ FragBF load_frag_bf16(const bf16_t* p) {
    FragBF f;
    f.q[0] = *(const uint4*)(p);
    f.q[1] = *(const uint4*)(p + 16);
    return f;
}

__device__ __forceinline__ v8f wmma_bf16(const FragBF& a, const FragBF& b, v8f c) {
    return __builtin_amdgcn_wmma_f32_16x16x32_bf16(false, a.v, false, b.v,
                                                   (short)0, c, false, false);
}

// ---------------------------------------------------------------------------
// f32 -> bf16 bulk convert: 8 elements/thread, packed b128 store.
// ---------------------------------------------------------------------------
__global__ __launch_bounds__(256) void cvt_bf16_kernel(const float* __restrict__ in,
                                                       bf16_t* __restrict__ out) {
    const size_t idx = ((size_t)blockIdx.x * 256 + threadIdx.x) * 8;
    const float4 a = *(const float4*)(in + idx);
    const float4 b = *(const float4*)(in + idx + 4);
    const float tmp[8] = {a.x, a.y, a.z, a.w, b.x, b.y, b.z, b.w};
    PackBF8 pk;
#pragma unroll
    for (int j = 0; j < 8; ++j) pk.e[j] = (bf16_t)tmp[j];
    *(uint4*)(out + idx) = pk.q;
}

// ---------------------------------------------------------------------------
// Linear: out = X @ W^T + bias, X:[4096,1024] bf16, W:[1024,1024] bf16, K=1024.
// One wave -> 64 n x 32 m output. block=256 (8 waves), grid=256.
// MODE 0: C^T orient, out = Q bf16 [B,H,L,64], scaled by 0.125
// MODE 1: C^T orient, out = K bf16 [B,H,L,64]
// MODE 2: C   orient, out = V^T bf16 [B,H,64,L]
// MODE 3: C^T orient, out = f32 [B*L, 1024]
// ---------------------------------------------------------------------------
template <int MODE>
__global__ __launch_bounds__(256) void linear_kernel(const bf16_t* __restrict__ X,
                                                     const bf16_t* __restrict__ W,
                                                     const float* __restrict__ bias,
                                                     void* __restrict__ out_) {
    const int K    = 1024;
    const int lane = threadIdx.x & 31;
    const int wid  = threadIdx.x >> 5;
    const int gw   = blockIdx.x * 8 + wid;   // 0..2047
    const int mgrp   = gw & 127;             // 32-token group
    const int ngroup = gw >> 7;              // 0..15 (64-col group == head)
    const int half  = lane >> 4;
    const int r     = lane & 15;
    const int kbase = half * 8;

    v8f c[4][2];
#pragma unroll
    for (int t = 0; t < 4; ++t)
#pragma unroll
        for (int u = 0; u < 2; ++u)
#pragma unroll
            for (int i = 0; i < 8; ++i) c[t][u][i] = 0.0f;

    for (int k0 = 0; k0 < K; k0 += 32) {
        FragBF wf[4], xf[2];
#pragma unroll
        for (int t = 0; t < 4; ++t)
            wf[t] = load_frag_bf16(W + (size_t)(ngroup * 64 + t * 16 + r) * K + k0 + kbase);
#pragma unroll
        for (int u = 0; u < 2; ++u)
            xf[u] = load_frag_bf16(X + (size_t)(mgrp * 32 + u * 16 + r) * K + k0 + kbase);
#pragma unroll
        for (int t = 0; t < 4; ++t)
#pragma unroll
            for (int u = 0; u < 2; ++u) {
                if constexpr (MODE == 2)
                    c[t][u] = wmma_bf16(xf[u], wf[t], c[t][u]);   // C: row=m, col=n
                else
                    c[t][u] = wmma_bf16(wf[t], xf[u], c[t][u]);   // C^T: row=n, col=m
            }
    }

    if constexpr (MODE == 2) {
        // C orientation: lane col n fixed, rows m = i (consecutive) -> pack tokens.
#pragma unroll
        for (int t = 0; t < 4; ++t) {
            const int n  = ngroup * 64 + t * 16 + r;     // h = ngroup, d = t*16+r
            const float bv = bias[n];
#pragma unroll
            for (int u = 0; u < 2; ++u) {
                const int mbase = mgrp * 32 + u * 16 + 8 * half;
                const int bb = mbase >> 11;
                const int l  = mbase & 2047;
                PackBF8 pk;
#pragma unroll
                for (int i = 0; i < 8; ++i) pk.e[i] = (bf16_t)(c[t][u][i] + bv);
                bf16_t* dst = (bf16_t*)out_ +
                    (((size_t)bb * 16 + ngroup) * 64 + (t * 16 + r)) * 2048 + l;
                *(uint4*)dst = pk.q;
            }
        }
    } else {
        // C^T orientation: lane col m fixed, rows n = i (consecutive) -> pack dims.
#pragma unroll
        for (int t = 0; t < 4; ++t) {
            const int nbase = ngroup * 64 + t * 16 + 8 * half;
            const float4 b0 = *(const float4*)(bias + nbase);
            const float4 b1 = *(const float4*)(bias + nbase + 4);
            const float bv[8] = {b0.x, b0.y, b0.z, b0.w, b1.x, b1.y, b1.z, b1.w};
#pragma unroll
            for (int u = 0; u < 2; ++u) {
                const int m  = mgrp * 32 + u * 16 + r;   // token
                if constexpr (MODE == 3) {
                    float* dst = (float*)out_ + (size_t)m * 1024 + nbase;
                    float v[8];
#pragma unroll
                    for (int i = 0; i < 8; ++i) v[i] = c[t][u][i] + bv[i];
                    *(float4*)(dst)     = make_float4(v[0], v[1], v[2], v[3]);
                    *(float4*)(dst + 4) = make_float4(v[4], v[5], v[6], v[7]);
                } else {
                    const int bb = m >> 11;
                    const int l  = m & 2047;
                    PackBF8 pk;
#pragma unroll
                    for (int i = 0; i < 8; ++i) {
                        float v = c[t][u][i] + bv[i];
                        if constexpr (MODE == 0) v *= 0.125f;   // fold 1/sqrt(d_k)
                        pk.e[i] = (bf16_t)v;
                    }
                    bf16_t* dst = (bf16_t*)out_ +
                        (((size_t)bb * 16 + ngroup) * 2048 + l) * 64 + t * 16 + 8 * half;
                    *(uint4*)dst = pk.q;
                }
            }
        }
    }
}

// ---------------------------------------------------------------------------
// Fused causal attention (flash style), transposed-score formulation.
// S^T = K_blk @ Q^T  -> each lane's 16 C-elements all belong to one query,
// softmax is in-lane + one shfl_xor(16); the exp'd values are already laid
// out as the B-fragment of P^T, so O^T = V^T @ P^T with no transpose at all.
// One wave per 16-row Q tile; key blocks of 32; d_k = 64.
// block = 256 (8 waves), grid = 512 (B*H*L/16 = 4096 waves).
// ---------------------------------------------------------------------------
__global__ __launch_bounds__(256) void attn_kernel(const bf16_t* __restrict__ Q,
                                                   const bf16_t* __restrict__ Km,
                                                   const bf16_t* __restrict__ Vt,
                                                   bf16_t* __restrict__ ctx) {
    const int lane = threadIdx.x & 31;
    const int wid  = threadIdx.x >> 5;
    const int gw   = blockIdx.x * 8 + wid;   // 0..4095
    const int bh    = gw >> 7;               // b*16 + h
    const int qtile = gw & 127;
    const int half  = lane >> 4;
    const int r     = lane & 15;
    const int kbase = half * 8;

    const bf16_t* Qb = Q  + (size_t)bh * 2048 * 64;
    const bf16_t* Kb = Km + (size_t)bh * 2048 * 64;
    const bf16_t* Vb = Vt + (size_t)bh * 64 * 2048;

    const int q = qtile * 16 + r;            // this lane's query (column of S^T)
    FragBF qf0 = load_frag_bf16(Qb + (size_t)q * 64 + kbase);        // B-frag of Q^T
    FragBF qf1 = load_frag_bf16(Qb + (size_t)q * 64 + 32 + kbase);

    v8f o[4];
#pragma unroll
    for (int t = 0; t < 4; ++t)
#pragma unroll
        for (int i = 0; i < 8; ++i) o[t][i] = 0.0f;
    float m_run = -3.0e38f;
    float l_run = 0.0f;

    const int nblk = (qtile * 16 + 15) / 32 + 1;   // causal: blocks up to diagonal

    for (int kb = 0; kb < nblk; ++kb) {
        // ---- S^T = K_blk @ Q^T  (two 16x16 C tiles; rows=keys, col=query) ----
        v8f st[2];
#pragma unroll
        for (int t = 0; t < 2; ++t) {
#pragma unroll
            for (int i = 0; i < 8; ++i) st[t][i] = 0.0f;
            const bf16_t* kp = Kb + (size_t)(kb * 32 + t * 16 + r) * 64 + kbase;
            FragBF ka = load_frag_bf16(kp);
            FragBF kb2 = load_frag_bf16(kp + 32);
            st[t] = wmma_bf16(ka, qf0, st[t]);
            st[t] = wmma_bf16(kb2, qf1, st[t]);
        }

        if (kb + 1 < nblk) {   // hint next K/V block toward the caches
            __builtin_prefetch(Kb + (size_t)((kb + 1) * 32 + lane) * 64, 0, 1);
            __builtin_prefetch(Vb + (size_t)lane * 2048 + (kb + 1) * 32, 0, 1);
            __builtin_prefetch(Vb + (size_t)(lane + 32) * 2048 + (kb + 1) * 32, 0, 1);
        }

        // ---- causal mask (Q was pre-scaled by 0.125 in its projection) ----
#pragma unroll
        for (int t = 0; t < 2; ++t)
#pragma unroll
            for (int i = 0; i < 8; ++i) {
                const int kk = kb * 32 + t * 16 + i + 8 * half;
                if (kk > q) st[t][i] += NEG_INF_MASK;
            }

        // ---- online softmax: in-lane over 16 elems + one cross-half swap ----
        float mx = st[0][0];
#pragma unroll
        for (int i = 1; i < 8; ++i) mx = fmaxf(mx, st[0][i]);
#pragma unroll
        for (int i = 0; i < 8; ++i) mx = fmaxf(mx, st[1][i]);
        mx = fmaxf(mx, __shfl_xor(mx, 16, 32));
        const float m_new = fmaxf(m_run, mx);
        const float alpha = __expf(m_run - m_new);
        m_run = m_new;

        float su = 0.0f;
#pragma unroll
        for (int t = 0; t < 2; ++t)
#pragma unroll
            for (int i = 0; i < 8; ++i) {
                const float p = __expf(st[t][i] - m_new);
                st[t][i] = p;
                su += p;
            }
        su += __shfl_xor(su, 16, 32);
        l_run = l_run * alpha + su;

#pragma unroll
        for (int t = 0; t < 4; ++t)
#pragma unroll
            for (int i = 0; i < 8; ++i) o[t][i] *= alpha;

        // ---- P^T B-fragment straight from the S^T C-fragments ----
        FragBF pf;
#pragma unroll
        for (int j = 0; j < 8; ++j) {
            pf.e[j]     = (bf16_t)st[0][j];
            pf.e[j + 8] = (bf16_t)st[1][j];
        }

        // ---- O^T += V^T_blk @ P^T  (A = V^T rows, row-major loads) ----
#pragma unroll
        for (int t = 0; t < 4; ++t) {
            FragBF vf = load_frag_bf16(Vb + (size_t)(t * 16 + r) * 2048 + kb * 32 + kbase);
            o[t] = wmma_bf16(vf, pf, o[t]);
        }
    }

    // ---- epilogue: O^T lane holds 8 consecutive dims per tile -> b128 stores ----
    const int bb = bh >> 4;
    const int h  = bh & 15;
    const float inv = 1.0f / l_run;
    bf16_t* crow = ctx + ((size_t)(bb * 2048 + qtile * 16 + r)) * 1024 + h * 64;
#pragma unroll
    for (int t = 0; t < 4; ++t) {
        PackBF8 pk;
#pragma unroll
        for (int i = 0; i < 8; ++i) pk.e[i] = (bf16_t)(o[t][i] * inv);
        *(uint4*)(crow + t * 16 + 8 * half) = pk.q;
    }
}

// ---------------------------------------------------------------------------
extern "C" void kernel_launch(void* const* d_in, const int* in_sizes, int n_in,
                              void* d_out, int out_size, void* d_ws, size_t ws_size,
                              hipStream_t stream) {
    const float* query = (const float*)d_in[0];
    const float* w_q   = (const float*)d_in[1];
    const float* b_q   = (const float*)d_in[2];
    const float* w_k   = (const float*)d_in[3];
    const float* b_k   = (const float*)d_in[4];
    const float* w_v   = (const float*)d_in[5];
    const float* b_v   = (const float*)d_in[6];
    const float* w_o   = (const float*)d_in[7];
    const float* b_o   = (const float*)d_in[8];

    const size_t NTOK = (size_t)4096 * 1024;   // 4,194,304
    const size_t NW   = (size_t)1024 * 1024;   // 1,048,576

    bf16_t* qbf = (bf16_t*)d_ws;
    bf16_t* wqb = qbf + NTOK;
    bf16_t* wkb = wqb + NW;
    bf16_t* wvb = wkb + NW;
    bf16_t* wob = wvb + NW;
    bf16_t* Qh  = wob + NW;    // [B,H,L,64], pre-scaled by 0.125
    bf16_t* Kh  = Qh + NTOK;   // [B,H,L,64]
    bf16_t* Vth = Kh + NTOK;   // [B,H,64,L]
    bf16_t* ctx = Vth + NTOK;  // [B*L, 1024]

    cvt_bf16_kernel<<<2048, 256, 0, stream>>>(query, qbf);
    cvt_bf16_kernel<<<512, 256, 0, stream>>>(w_q, wqb);
    cvt_bf16_kernel<<<512, 256, 0, stream>>>(w_k, wkb);
    cvt_bf16_kernel<<<512, 256, 0, stream>>>(w_v, wvb);
    cvt_bf16_kernel<<<512, 256, 0, stream>>>(w_o, wob);

    linear_kernel<0><<<256, 256, 0, stream>>>(qbf, wqb, b_q, (void*)Qh);
    linear_kernel<1><<<256, 256, 0, stream>>>(qbf, wkb, b_k, (void*)Kh);
    linear_kernel<2><<<256, 256, 0, stream>>>(qbf, wvb, b_v, (void*)Vth);

    attn_kernel<<<512, 256, 0, stream>>>(Qh, Kh, Vth, ctx);

    linear_kernel<3><<<256, 256, 0, stream>>>(ctx, wob, b_o, d_out);
}
